// LSTM_40793599377518
// MI455X (gfx1250) — compile-verified, matrix operand order checked by reading, not run on hardware
//
#include <hip/hip_runtime.h>
#include <hip/hip_bf16.h>
#include <stdint.h>
#include <stddef.h>

// ---------------------------------------------------------------------------
// LSTM on MI455X (gfx1250): bf16 WMMA GEMMs with fp32 accumulation.
//   Phase 1: xw = x @ w + b           (16384 x 1024 x 4096, parallel)
//   Phase 2: 512 sequential steps of  act = xw[t] + h @ recur_w, cell update
//            (K split over 4 waves/WG, fp32 partials reduced through LDS)
// ---------------------------------------------------------------------------

typedef __attribute__((ext_vector_type(16))) __bf16 v16bf;
typedef __attribute__((ext_vector_type(8)))  __bf16 v8bf;
typedef __attribute__((ext_vector_type(8)))  float  v8f;

#define B_   32
#define T_   512
#define F_   1024
#define H_   1024
#define G4H  4096
#define KT_F 32    // F_/32 k-tiles
#define KT_H 32    // H_/32 k-tiles
#define SKW  4     // waves splitting K in the step kernel

__device__ __forceinline__ v8f wmma_bf16(v16bf a, v16bf b, v8f c) {
  // v_wmma_f32_16x16x32_bf16: D = A(16x32) * B(32x16) + C(16x16 f32)
  return __builtin_amdgcn_wmma_f32_16x16x32_bf16(
      /*neg_a=*/false, a, /*neg_b=*/false, b,
      /*c_mod=*/(short)0, c, /*reuse_a=*/false, /*reuse_b=*/false);
}

__device__ __forceinline__ float sigf(float x) {
  return 1.0f / (1.0f + __expf(-x));
}

// ---------------------------------------------------------------------------
// fp32 -> bf16 row-major convert (for A-operands: x)
// ---------------------------------------------------------------------------
__global__ void convert_bf16_kernel(const float* __restrict__ src,
                                    __bf16* __restrict__ dst, int n) {
  int i = blockIdx.x * blockDim.x + threadIdx.x;
  if (i < n) dst[i] = (__bf16)src[i];
}

// ---------------------------------------------------------------------------
// Pack a row-major [K=1024, N=4096] fp32 matrix into the WMMA B-operand
// layout: [ntile][ktile][lane][8 x (2 bf16)], per-lane contiguous 32B.
//   lane: n = lane&15, khalf = lane>>4
//   word v: kk = (v<4 ? 2v + 8*khalf : 16 + 2(v-4) + 8*khalf), pair (kk,kk+1)
// ---------------------------------------------------------------------------
__global__ void pack_b_kernel(const float* __restrict__ src,
                              __bf16* __restrict__ dst) {
  int gid  = blockIdx.x * blockDim.x + threadIdx.x;   // 2^21 words total
  int v    = gid & 7;
  int lane = (gid >> 3) & 31;
  int kt   = (gid >> 8) & 31;
  int nt   = gid >> 13;                               // 0..255
  int n    = nt * 16 + (lane & 15);
  int kh   = lane >> 4;
  int kk   = (v < 4) ? (v * 2 + kh * 8) : (16 + (v - 4) * 2 + kh * 8);
  int k    = kt * 32 + kk;
  dst[(size_t)gid * 2 + 0] = (__bf16)src[(size_t)k * G4H + n];
  dst[(size_t)gid * 2 + 1] = (__bf16)src[(size_t)(k + 1) * G4H + n];
}

// ---------------------------------------------------------------------------
// Zero the recurrent state: h (bf16 buf 0), h (fp32), cell state s (fp32)
// ---------------------------------------------------------------------------
__global__ void init_state_kernel(__bf16* __restrict__ hbf0,
                                  float* __restrict__ hf,
                                  float* __restrict__ s) {
  int i = blockIdx.x * blockDim.x + threadIdx.x;      // 32*1024 elements
  hbf0[i] = (__bf16)0.0f;
  hf[i]   = 0.0f;
  s[i]    = 0.0f;
}

// ---------------------------------------------------------------------------
// Phase 1 GEMM: xw[16384,4096] = x_bf[16384,1024] @ wB + bias
// One wave computes a 32(M) x 64(N) tile: 2 M-tiles x 4 N-tiles, K=1024.
// The 8 waves of a block share the same M rows -> A tiles hit WGP$.
// ---------------------------------------------------------------------------
__global__ __launch_bounds__(256) void gemm_xw_kernel(
    const __bf16* __restrict__ xbf, const __bf16* __restrict__ wB,
    const float* __restrict__ bias, float* __restrict__ xw) {
  const int lane = threadIdx.x & 31;
  const int wave = threadIdx.x >> 5;
  const int tile = blockIdx.x * 8 + wave;   // 32768 wave-tiles
  const int tM   = tile >> 6;               // 512 M-tiles of 32 rows
  const int tN   = tile & 63;               // 64 col-groups of 64
  const int n    = lane & 15;
  const int kh   = lane >> 4;

  v8f acc[4][2];
#pragma unroll
  for (int i = 0; i < 4; ++i)
#pragma unroll
    for (int j = 0; j < 2; ++j) acc[i][j] = (v8f){};

  for (int kt = 0; kt < KT_F; ++kt) {
    v16bf A[2];
#pragma unroll
    for (int mt = 0; mt < 2; ++mt) {
      const __bf16* ap =
          xbf + (size_t)(tM * 32 + mt * 16 + n) * F_ + kt * 32 + kh * 8;
      v8bf lo = *(const v8bf*)ap;          // K = kh*8 .. +7
      v8bf hi = *(const v8bf*)(ap + 16);   // K = 16+kh*8 .. +7
      A[mt] = __builtin_shufflevector(lo, hi, 0, 1, 2, 3, 4, 5, 6, 7, 8, 9, 10,
                                      11, 12, 13, 14, 15);
    }
#pragma unroll
    for (int nt = 0; nt < 4; ++nt) {
      const int ntile = tN * 4 + nt;
      v16bf Bm =
          *(const v16bf*)(wB + (((size_t)ntile * KT_F + kt) * 32 + lane) * 16);
      acc[nt][0] = wmma_bf16(A[0], Bm, acc[nt][0]);
      acc[nt][1] = wmma_bf16(A[1], Bm, acc[nt][1]);
    }
  }
#pragma unroll
  for (int nt = 0; nt < 4; ++nt) {
    const int col = (tN * 4 + nt) * 16 + n;
    const float bb = bias[col];
#pragma unroll
    for (int mt = 0; mt < 2; ++mt) {
#pragma unroll
      for (int v = 0; v < 8; ++v) {
        const int m = tM * 32 + mt * 16 + kh * 8 + v;
        xw[(size_t)m * G4H + col] = acc[nt][mt][v] + bb;
      }
    }
  }
}

// ---------------------------------------------------------------------------
// Phase 2: one LSTM step. 64 workgroups x 4 waves; WG `wg` owns output
// columns j = wg*16 .. +15 of H. Each wave accumulates 8 of the 32 k-tiles
// for all 4 gates x 2 M-tiles; partials are summed through LDS, then the
// cell update is spread over all 128 threads. h bf16 ping-pongs per step.
// ---------------------------------------------------------------------------
__global__ __launch_bounds__(128) void lstm_step_kernel(
    const float* __restrict__ xw,        // [B*T, 4H], row = m*T_ + t
    const __bf16* __restrict__ rwB,      // packed B layout [256][32][32][16]
    const __bf16* __restrict__ hbf_in,   // [32, 1024] row-major
    __bf16* __restrict__ hbf_out,        // [32, 1024] row-major
    float* __restrict__ hf,              // [32, 1024] fp32 h state
    float* __restrict__ s,               // [32, 1024] fp32 cell state
    const unsigned char* __restrict__ mask,  // [B*T] bool
    float* __restrict__ out,             // [B, T, H]
    int t) {
  const int tid  = threadIdx.x;
  const int lane = tid & 31;
  const int wv   = tid >> 5;        // 0..3: K-split wave id
  const int wg   = blockIdx.x;      // 0..63: 16-column tile of H
  const int n    = lane & 15;
  const int kh   = lane >> 4;

  // red[wv][g][mt][lane][v] : 4*4*2*32*8 floats = 32 KB
  __shared__ float red[SKW * 4 * 2 * 32 * 8];

  v8f acc[4][2];
#pragma unroll
  for (int g = 0; g < 4; ++g)
#pragma unroll
    for (int mt = 0; mt < 2; ++mt) acc[g][mt] = (v8f){};

  const int kt0 = wv * (KT_H / SKW);
  for (int kt = kt0; kt < kt0 + (KT_H / SKW); ++kt) {
    v16bf A[2];
#pragma unroll
    for (int mt = 0; mt < 2; ++mt) {
      const __bf16* ap =
          hbf_in + (size_t)(mt * 16 + n) * H_ + kt * 32 + kh * 8;
      v8bf lo = *(const v8bf*)ap;
      v8bf hi = *(const v8bf*)(ap + 16);
      A[mt] = __builtin_shufflevector(lo, hi, 0, 1, 2, 3, 4, 5, 6, 7, 8, 9, 10,
                                      11, 12, 13, 14, 15);
    }
#pragma unroll
    for (int g = 0; g < 4; ++g) {
      const int ntile = g * 64 + wg;   // column tile in 4H
      v16bf Bm =
          *(const v16bf*)(rwB + (((size_t)ntile * KT_H + kt) * 32 + lane) * 16);
      acc[g][0] = wmma_bf16(A[0], Bm, acc[g][0]);
      acc[g][1] = wmma_bf16(A[1], Bm, acc[g][1]);
    }
  }

  // Dump per-wave partial accumulators into LDS (contiguous 8 floats/lane).
#pragma unroll
  for (int g = 0; g < 4; ++g) {
#pragma unroll
    for (int mt = 0; mt < 2; ++mt) {
      float* p = &red[((((wv * 4 + g) * 2 + mt) * 32) + lane) * 8];
#pragma unroll
      for (int v = 0; v < 8; ++v) p[v] = acc[g][mt][v];
    }
  }
  __syncthreads();

  // Reduce over the 4 K-split waves + LSTM cell update; 4 elements/thread.
#pragma unroll
  for (int e4 = 0; e4 < 4; ++e4) {
    const int e      = tid * 4 + e4;      // 0..511: (mt, v, lane) element
    const int lane_e = e & 31;
    const int v_e    = (e >> 5) & 7;
    const int mt_e   = e >> 8;
    float a[4];
#pragma unroll
    for (int g = 0; g < 4; ++g) {
      float sum = 0.0f;
#pragma unroll
      for (int w = 0; w < SKW; ++w)
        sum += red[((((w * 4 + g) * 2 + mt_e) * 32) + lane_e) * 8 + v_e];
      a[g] = sum;
    }
    const int m   = mt_e * 16 + (lane_e >> 4) * 8 + v_e;   // batch row 0..31
    const int col = wg * 16 + (lane_e & 15);               // column in H
    const size_t row = (size_t)m * T_ + t;                 // [B,T] flat index
    const float a1 = a[0] + xw[row * G4H + 0 * H_ + col];
    const float a2 = a[1] + xw[row * G4H + 1 * H_ + col];
    const float a3 = a[2] + xw[row * G4H + 2 * H_ + col];
    const float a4 = a[3] + xw[row * G4H + 3 * H_ + col];
    const int sidx = m * H_ + col;
    const float sold = s[sidx];
    const float snew = sigf(a2) * sold + sigf(a1) * tanhf(a3);
    const float hnew = tanhf(snew) * sigf(a4);
    const bool mk = mask[row] != 0;
    const float hv = mk ? hnew : hf[sidx];
    const float sv = mk ? snew : sold;
    s[sidx]  = sv;
    hf[sidx] = hv;
    hbf_out[sidx] = (__bf16)hv;
    out[row * (size_t)H_ + col] = hv;
  }
}

// ---------------------------------------------------------------------------
// Host launcher. Inputs: x[B,T,F] f32, mask[B,T,1] bool, w[F,4H] f32,
// recur_w[H,4H] f32, b[1,4H] f32. Output: [B,T,H] f32.
// ---------------------------------------------------------------------------
extern "C" void kernel_launch(void* const* d_in, const int* in_sizes, int n_in,
                              void* d_out, int out_size, void* d_ws,
                              size_t ws_size, hipStream_t stream) {
  const float* x          = (const float*)d_in[0];
  const unsigned char* mk = (const unsigned char*)d_in[1];
  const float* w          = (const float*)d_in[2];
  const float* rw         = (const float*)d_in[3];
  const float* b          = (const float*)d_in[4];
  float* out              = (float*)d_out;

  char* ws = (char*)d_ws;
  // Workspace layout (bytes):
  __bf16* xbf  = (__bf16*)(ws + 0);            //  33,554,432  x as bf16
  __bf16* wB   = (__bf16*)(ws + 33554432);     //   8,388,608  packed w
  __bf16* rwB  = (__bf16*)(ws + 41943040);     //   8,388,608  packed recur_w
  float*  xw   = (float*)(ws + 50331648);      // 268,435,456  x@w + b
  __bf16* hbf0 = (__bf16*)(ws + 318767104);    //      65,536  h bf16 buf0
  __bf16* hbf1 = (__bf16*)(ws + 318832640);    //      65,536  h bf16 buf1
  float*  hf   = (float*)(ws + 318898176);     //     131,072  h fp32
  float*  s    = (float*)(ws + 319029248);     //     131,072  cell state
  (void)in_sizes; (void)n_in; (void)out_size; (void)ws_size;

  // One-time conversions / packs.
  convert_bf16_kernel<<<65536, 256, 0, stream>>>(x, xbf, B_ * T_ * F_);
  pack_b_kernel<<<8192, 256, 0, stream>>>(w, wB);
  pack_b_kernel<<<8192, 256, 0, stream>>>(rw, rwB);
  init_state_kernel<<<128, 256, 0, stream>>>(hbf0, hf, s);

  // Phase 1: big GEMM.
  gemm_xw_kernel<<<4096, 256, 0, stream>>>(xbf, wB, b, xw);

  // Phase 2: sequential recurrence, one launch per step (kernel boundary is
  // the grid-wide sync; h bf16 buffer ping-pongs to avoid RW races).
  for (int t = 0; t < T_; ++t) {
    const __bf16* hin = (t & 1) ? hbf1 : hbf0;
    __bf16* hout      = (t & 1) ? hbf0 : hbf1;
    lstm_step_kernel<<<64, 128, 0, stream>>>(xw, rwB, hin, hout, hf, s, mk, out,
                                             t);
  }
}
